// SwinBlock_26551487824137
// MI455X (gfx1250) — matrix operand order repaired and checked by reading, not compiled, not run on hardware
//
#include <hip/hip_runtime.h>
#include <hip/hip_bf16.h>

// ---------------------------------------------------------------------------
// Swin block for gfx1250 (MI455X): all GEMMs via v_wmma_f32_16x16x32_bf16,
// GEMM staging via GLOBAL_LOAD_ASYNC_TO_LDS_B128 (ASYNCcnt) + double buffer.
// ---------------------------------------------------------------------------

#define DIM      384
#define WS_      8
#define SHIFT_   4
#define HEADS_   8
#define HDIM     48
#define NTOK     32768            // 8 * 64 * 64
#define NWIN     512              // 8 * 8 * 8
#define SCALE_   0.14433756729740643f   // 48^-0.5
#define QK_ELEMS (512u*8u*64u*48u)      // 12,582,912 per q/k/vT buffer

typedef __bf16 bf16_t;
typedef __attribute__((ext_vector_type(8)))  __bf16 v8bf;
typedef __attribute__((ext_vector_type(16))) __bf16 v16bf;
typedef __attribute__((ext_vector_type(8)))  float  v8f;

__device__ __forceinline__ bf16_t f2bf(float f) {
    unsigned int u = __builtin_bit_cast(unsigned int, f);
    u += 0x7FFFu + ((u >> 16) & 1u);                 // round-to-nearest-even
    unsigned short s = (unsigned short)(u >> 16);
    return __builtin_bit_cast(bf16_t, s);
}
__device__ __forceinline__ bf16_t bf_zero() {
    return __builtin_bit_cast(bf16_t, (unsigned short)0);
}

// Async 16B copy global -> LDS (per-lane), tracked by ASYNCcnt.
// LDS address = low 32 bits of the generic pointer (ISA: LDS aperture keeps
// the wave-relative LDS byte address in addr[31:0]).
__device__ __forceinline__ void async_cp16(const bf16_t* g, bf16_t* l) {
    unsigned int       laddr = (unsigned int)(unsigned long long)l;
    unsigned long long gaddr = (unsigned long long)g;
    asm volatile("global_load_async_to_lds_b128 %0, %1, off"
                 :: "v"(laddr), "v"(gaddr) : "memory");
}
__device__ __forceinline__ void wait_async0() {
    asm volatile("s_wait_asynccnt 0x0" ::: "memory");
}

// A-fragment (16x32 bf16, MxK): lane holds row m = lane%16.
// element e -> K = (e<8 ? e : e+8) + 8*(lane>=16)   [ISA 7.12.2]
__device__ __forceinline__ v16bf make_afrag(const bf16_t* row, int half) {
    v8bf lo = *(const v8bf*)(row + 8 * half);
    v8bf hi = *(const v8bf*)(row + 16 + 8 * half);
    v16bf a;
#pragma unroll
    for (int i = 0; i < 8; i++) { a[i] = lo[i]; a[8 + i] = hi[i]; }
    return a;
}
// B-fragment (32x16 bf16, KxN) fed from Wt stored [n][k]: lane holds col
// n = lane%16; element e -> K = (e<8 ? e : e+8) + 8*(lane>=16)
__device__ __forceinline__ v16bf make_bfrag(const bf16_t* row, int half) {
    v8bf lo = *(const v8bf*)(row + 8 * half);
    v8bf hi = *(const v8bf*)(row + 16 + 8 * half);
    v16bf b;
#pragma unroll
    for (int i = 0; i < 8; i++) { b[i] = lo[i]; b[8 + i] = hi[i]; }
    return b;
}

// ---------------------------------------------------------------------------
// Weight transpose + fp32->bf16:  wt[n*K + k] = bf16(w[k*N + n])
// ---------------------------------------------------------------------------
__global__ __launch_bounds__(256) void wconv_kernel(
    const float* __restrict__ w, bf16_t* __restrict__ wt, int K, int N) {
    int idx = blockIdx.x * 256 + threadIdx.x;
    if (idx >= K * N) return;
    int n = idx / K, k = idx - n * K;
    wt[idx] = f2bf(w[(size_t)k * N + n]);
}

// ---------------------------------------------------------------------------
// LN1 fused with shift + window partition.  One wave32 per token.
// Output rows are in window order: row = win*64 + t, win = b*64 + wh*8 + ww.
// ---------------------------------------------------------------------------
__global__ __launch_bounds__(256) void ln1_window_kernel(
    const float* __restrict__ x, const float* __restrict__ g,
    const float* __restrict__ bta, bf16_t* __restrict__ out) {
    int token = blockIdx.x * 8 + (threadIdx.x >> 5);
    int lane  = threadIdx.x & 31;
    int win = token >> 6, t = token & 63;
    int b  = win >> 6, wh = (win >> 3) & 7, ww = win & 7;
    int r  = t >> 3,  c  = t & 7;
    int hh = (wh * 8 + r + SHIFT_) & 63;
    int wc = (ww * 8 + c + SHIFT_) & 63;
    const float* xb = x + (size_t)b * DIM * 4096 + hh * 64 + wc;

    float vals[12], s = 0.f, s2 = 0.f;
#pragma unroll
    for (int j = 0; j < 12; j++) {
        int ch = lane + 32 * j;
        float v = xb[(size_t)ch * 4096];
        vals[j] = v; s += v; s2 += v * v;
    }
#pragma unroll
    for (int off = 1; off < 32; off <<= 1) {
        s  += __shfl_xor(s,  off, 32);
        s2 += __shfl_xor(s2, off, 32);
    }
    float mean = s * (1.f / DIM);
    float var  = s2 * (1.f / DIM) - mean * mean;
    float inv  = rsqrtf(var + 1e-5f);
    bf16_t* o = out + (size_t)token * DIM;
#pragma unroll
    for (int j = 0; j < 12; j++) {
        int ch = lane + 32 * j;
        o[ch] = f2bf((vals[j] - mean) * inv * g[ch] + bta[ch]);
    }
}

// ---------------------------------------------------------------------------
// LN2 over xh [32768, 384] fp32 (spatial row order) -> bf16
// ---------------------------------------------------------------------------
__global__ __launch_bounds__(256) void ln2_kernel(
    const float* __restrict__ xh, const float* __restrict__ g,
    const float* __restrict__ bta, bf16_t* __restrict__ out) {
    int token = blockIdx.x * 8 + (threadIdx.x >> 5);
    int lane  = threadIdx.x & 31;
    const float* xb = xh + (size_t)token * DIM;
    float vals[12], s = 0.f, s2 = 0.f;
#pragma unroll
    for (int j = 0; j < 12; j++) {
        int ch = lane + 32 * j;
        float v = xb[ch];
        vals[j] = v; s += v; s2 += v * v;
    }
#pragma unroll
    for (int off = 1; off < 32; off <<= 1) {
        s  += __shfl_xor(s,  off, 32);
        s2 += __shfl_xor(s2, off, 32);
    }
    float mean = s * (1.f / DIM);
    float var  = s2 * (1.f / DIM) - mean * mean;
    float inv  = rsqrtf(var + 1e-5f);
    bf16_t* o = out + (size_t)token * DIM;
#pragma unroll
    for (int j = 0; j < 12; j++) {
        int ch = lane + 32 * j;
        o[ch] = f2bf((vals[j] - mean) * inv * g[ch] + bta[ch]);
    }
}

// ---------------------------------------------------------------------------
// Generic bf16 WMMA GEMM: C[M,N] = A[M,K] @ Wt[N,K]^T  (+bias, mode epilogue)
// Block tile 128x128, BK=32 (one WMMA K-step). 256 threads = 8 waves,
// wave grid 4(M) x 2(N), each wave: 2x4 fragments of 16x16.
// LDS double-buffered, filled with GLOBAL_LOAD_ASYNC_TO_LDS_B128.
// ---------------------------------------------------------------------------
#define MODE_QKV  0
#define MODE_PROJ 1
#define MODE_GELU 2
#define MODE_OUT  3

__global__ __launch_bounds__(256) void gemm_bf16_kernel(
    const bf16_t* __restrict__ A, const bf16_t* __restrict__ Wt,
    const float* __restrict__ bias, int M, int N, int K, int mode,
    bf16_t* __restrict__ outb, float* __restrict__ outf,
    const float* __restrict__ xres) {
    __shared__ bf16_t As[2][128][32];
    __shared__ bf16_t Bs[2][128][32];

    const int tid  = threadIdx.x;
    const int lane = tid & 31, wid = tid >> 5;
    const int half = lane >> 4, l16 = lane & 15;
    const int wm = wid & 3, wn = wid >> 2;
    const int tile_m = blockIdx.y * 128, tile_n = blockIdx.x * 128;

    // this thread's two (row, 8-elem-offset) chunks within a 128x32 tile
    const int r0 = tid >> 2,         o0 = (tid & 3) * 8;
    const int r1 = (tid + 256) >> 2, o1 = ((tid + 256) & 3) * 8;

    v8f acc[2][4];
    v8f vz = {};
#pragma unroll
    for (int i = 0; i < 2; i++)
#pragma unroll
        for (int j = 0; j < 4; j++) acc[i][j] = vz;

    const int nt = K / 32;

    // prologue: async-fill buffer 0 with k-tile 0
    async_cp16(&A[(size_t)(tile_m + r0) * K + o0], &As[0][r0][o0]);
    async_cp16(&A[(size_t)(tile_m + r1) * K + o1], &As[0][r1][o1]);
    async_cp16(&Wt[(size_t)(tile_n + r0) * K + o0], &Bs[0][r0][o0]);
    async_cp16(&Wt[(size_t)(tile_n + r1) * K + o1], &Bs[0][r1][o1]);
    wait_async0();
    __syncthreads();

    for (int it = 0; it < nt; ++it) {
        const int cur = it & 1;
        if (it + 1 < nt) {            // async-fill the other buffer
            const int kt1 = (it + 1) * 32;
            async_cp16(&A[(size_t)(tile_m + r0) * K + kt1 + o0], &As[cur ^ 1][r0][o0]);
            async_cp16(&A[(size_t)(tile_m + r1) * K + kt1 + o1], &As[cur ^ 1][r1][o1]);
            async_cp16(&Wt[(size_t)(tile_n + r0) * K + kt1 + o0], &Bs[cur ^ 1][r0][o0]);
            async_cp16(&Wt[(size_t)(tile_n + r1) * K + kt1 + o1], &Bs[cur ^ 1][r1][o1]);
        }
        if (it + 2 < nt) {            // L2 prefetch two tiles ahead
            const int kt2 = (it + 2) * 32;
            __builtin_prefetch(&A[(size_t)(tile_m + r0) * K + kt2 + o0], 0, 1);
            __builtin_prefetch(&Wt[(size_t)(tile_n + r0) * K + kt2 + o0], 0, 1);
        }

        v16bf af[2], bfv[4];
#pragma unroll
        for (int fm = 0; fm < 2; fm++)
            af[fm] = make_afrag(&As[cur][wm * 32 + fm * 16 + l16][0], half);
#pragma unroll
        for (int fn = 0; fn < 4; fn++)
            bfv[fn] = make_bfrag(&Bs[cur][wn * 64 + fn * 16 + l16][0], half);
#pragma unroll
        for (int fm = 0; fm < 2; fm++)
#pragma unroll
            for (int fn = 0; fn < 4; fn++)
                acc[fm][fn] = __builtin_amdgcn_wmma_f32_16x16x32_bf16(
                    false, af[fm], false, bfv[fn], (short)0, acc[fm][fn],
                    false, false);

        wait_async0();     // this wave's fills for the next buffer landed
        __syncthreads();   // everyone done reading `cur`, next buffer ready
    }

    // Epilogue.  C element r of v8f: m = r + 8*half, n = lane%16.
#pragma unroll
    for (int fm = 0; fm < 2; fm++)
#pragma unroll
        for (int fn = 0; fn < 4; fn++)
#pragma unroll
            for (int r = 0; r < 8; r++) {
                int mg = tile_m + wm * 32 + fm * 16 + r + 8 * half;
                int ng = tile_n + wn * 64 + fn * 16 + l16;
                float v = acc[fm][fn][r] + bias[ng];
                if (mode == MODE_QKV) {
                    int which = ng / DIM, rem = ng % DIM;
                    int head = rem / HDIM, dim = rem % HDIM;
                    int win = mg >> 6, t = mg & 63;
                    if (which == 0) {
                        outb[((size_t)(win * 8 + head) * 64 + t) * HDIM + dim] =
                            f2bf(v * SCALE_);
                    } else if (which == 1) {
                        (outb + QK_ELEMS)[((size_t)(win * 8 + head) * 64 + t) * HDIM + dim] =
                            f2bf(v);
                    } else {  // v stored transposed: [win][head][dim][token]
                        (outb + 2u * QK_ELEMS)[((size_t)(win * 8 + head) * HDIM + dim) * 64 + t] =
                            f2bf(v);
                    }
                } else if (mode == MODE_PROJ) {
                    // window reverse + unshift + residual add, write xh fp32
                    int win = mg >> 6, t = mg & 63;
                    int b = win >> 6, wh = (win >> 3) & 7, ww = win & 7;
                    int rr = t >> 3, cc = t & 7;
                    int hh = (wh * 8 + rr + SHIFT_) & 63;
                    int wc = (ww * 8 + cc + SHIFT_) & 63;
                    v += xres[((size_t)(b * DIM + ng) * 64 + hh) * 64 + wc];
                    outf[(((size_t)b * 64 + hh) * 64 + wc) * DIM + ng] = v;
                } else if (mode == MODE_GELU) {
                    float gl = 0.5f * v * (1.f + erff(v * 0.70710678f));
                    outb[(size_t)mg * 1536 + ng] = f2bf(gl);
                } else {  // MODE_OUT: scatter to [B,C,H,W] fp32
                    int b = mg >> 12, hh = (mg >> 6) & 63, wc = mg & 63;
                    outf[((size_t)(b * DIM + ng) * 64 + hh) * 64 + wc] = v;
                }
            }
}

// ---------------------------------------------------------------------------
// Windowed attention: one block per (window, head).
// S = q@k^T (64x64, K=48 padded to 64), softmax f32, O = P@v (64x48, K=64).
// ---------------------------------------------------------------------------
__global__ __launch_bounds__(256) void attn_kernel(
    const bf16_t* __restrict__ q, const bf16_t* __restrict__ k,
    const bf16_t* __restrict__ vt, bf16_t* __restrict__ out) {
    __shared__ bf16_t qs[64][64];
    __shared__ bf16_t ks[64][64];
    __shared__ float  sb[64][64];
    __shared__ bf16_t ps[64][64];
    __shared__ bf16_t vs[48][64];

    int win = blockIdx.x >> 3, head = blockIdx.x & 7;
    size_t qkbase = (size_t)(win * 8 + head) * 64 * HDIM;

    bf16_t z = bf_zero();
#pragma unroll
    for (int i = 0; i < 16; i++) {
        int idx = threadIdx.x + 256 * i;
        int t = idx >> 6, d = idx & 63;
        qs[t][d] = (d < HDIM) ? q[qkbase + (size_t)t * HDIM + d] : z;
        ks[t][d] = (d < HDIM) ? k[qkbase + (size_t)t * HDIM + d] : z;
    }
#pragma unroll
    for (int i = 0; i < 12; i++) {
        int idx = threadIdx.x + 256 * i;
        vs[idx >> 6][idx & 63] = vt[qkbase + idx];
    }
    __syncthreads();

    int lane = threadIdx.x & 31, wid = threadIdx.x >> 5;
    int half = lane >> 4, l16 = lane & 15;

    // ---- S = q @ k^T : 16 fragments, 2 per wave ----
#pragma unroll
    for (int ff = 0; ff < 2; ff++) {
        int f = wid * 2 + ff;
        int fm = f >> 2, fn = f & 3;
        v8f acc = {};
#pragma unroll
        for (int kk = 0; kk < 64; kk += 32) {
            v16bf a = make_afrag(&qs[fm * 16 + l16][kk], half);
            v16bf b = make_bfrag(&ks[fn * 16 + l16][kk], half);
            acc = __builtin_amdgcn_wmma_f32_16x16x32_bf16(
                false, a, false, b, (short)0, acc, false, false);
        }
#pragma unroll
        for (int r = 0; r < 8; r++)
            sb[fm * 16 + r + 8 * half][fn * 16 + l16] = acc[r];
    }
    __syncthreads();

    // ---- softmax over rows (4 threads per row) ----
    {
        int row = threadIdx.x >> 2, quad = threadIdx.x & 3;
        float vc[16], mx = -1e30f;
#pragma unroll
        for (int i = 0; i < 16; i++) {
            float s = sb[row][quad * 16 + i];
            vc[i] = s; mx = fmaxf(mx, s);
        }
        mx = fmaxf(mx, __shfl_xor(mx, 1, 4));
        mx = fmaxf(mx, __shfl_xor(mx, 2, 4));
        float sum = 0.f;
#pragma unroll
        for (int i = 0; i < 16; i++) { vc[i] = __expf(vc[i] - mx); sum += vc[i]; }
        sum += __shfl_xor(sum, 1, 4);
        sum += __shfl_xor(sum, 2, 4);
        float rs = 1.f / sum;
#pragma unroll
        for (int i = 0; i < 16; i++) ps[row][quad * 16 + i] = f2bf(vc[i] * rs);
    }
    __syncthreads();

    // ---- O = P @ V : 12 fragments (4m x 3n), round-robin over waves ----
    for (int f = wid; f < 12; f += 8) {
        int fm = f / 3, fn = f % 3;
        v8f acc = {};
#pragma unroll
        for (int kk = 0; kk < 64; kk += 32) {
            v16bf a = make_afrag(&ps[fm * 16 + l16][kk], half);
            v16bf b = make_bfrag(&vs[fn * 16 + l16][kk], half);
            acc = __builtin_amdgcn_wmma_f32_16x16x32_bf16(
                false, a, false, b, (short)0, acc, false, false);
        }
#pragma unroll
        for (int r = 0; r < 8; r++) {
            int m = fm * 16 + r + 8 * half;
            int n = fn * 16 + l16;
            out[((size_t)(win * 64 + m)) * DIM + head * HDIM + n] = f2bf(acc[r]);
        }
    }
}

// ---------------------------------------------------------------------------
// Launcher
// ---------------------------------------------------------------------------
extern "C" void kernel_launch(void* const* d_in, const int* in_sizes, int n_in,
                              void* d_out, int out_size, void* d_ws, size_t ws_size,
                              hipStream_t stream) {
    const float* x      = (const float*)d_in[0];
    const float* qkv_w  = (const float*)d_in[1];
    const float* qkv_b  = (const float*)d_in[2];
    const float* proj_w = (const float*)d_in[3];
    const float* proj_b = (const float*)d_in[4];
    const float* ln1_g  = (const float*)d_in[5];
    const float* ln1_b  = (const float*)d_in[6];
    const float* ln2_g  = (const float*)d_in[7];
    const float* ln2_b  = (const float*)d_in[8];
    const float* mlp_w1 = (const float*)d_in[9];
    const float* mlp_b1 = (const float*)d_in[10];
    const float* mlp_w2 = (const float*)d_in[11];
    const float* mlp_b2 = (const float*)d_in[12];

    char* ws = (char*)d_ws;
    // weight region
    bf16_t* qkv_wT  = (bf16_t*)(ws + 0);                       // 1152*384
    bf16_t* proj_wT = (bf16_t*)(ws + 884736);                  // 384*384
    bf16_t* w1T     = (bf16_t*)(ws + 884736 + 294912);         // 1536*384
    bf16_t* w2T     = (bf16_t*)(ws + 884736 + 294912 + 1179648); // 384*1536
    // rotating arenas (R1 / R2), 112 MB each
    const size_t R1 = 4ull * 1024 * 1024;
    const size_t R2 = R1 + 117440512ull;
    bf16_t* A0       = (bf16_t*)(ws + R1);   // LN1 out   [32768,384]
    bf16_t* qbuf     = (bf16_t*)(ws + R2);   // q | k | vT (3 * QK_ELEMS)
    bf16_t* attn_out = (bf16_t*)(ws + R1);   // reuses A0 slot
    float*  xh       = (float*)(ws + R2);    // reuses q/k/vT slot
    bf16_t* hbuf     = (bf16_t*)(ws + R1);   // reuses attn_out slot
    bf16_t* hid      = (bf16_t*)(ws + R2);   // reuses xh slot [32768,1536]
    float*  outp     = (float*)d_out;

    // 1) weights -> bf16 transposed [N][K]
    wconv_kernel<<<(384 * 1152 + 255) / 256, 256, 0, stream>>>(qkv_w,  qkv_wT,  384, 1152);
    wconv_kernel<<<(384 * 384  + 255) / 256, 256, 0, stream>>>(proj_w, proj_wT, 384, 384);
    wconv_kernel<<<(384 * 1536 + 255) / 256, 256, 0, stream>>>(mlp_w1, w1T,     384, 1536);
    wconv_kernel<<<(1536 * 384 + 255) / 256, 256, 0, stream>>>(mlp_w2, w2T,     1536, 384);

    // 2) LN1 + shift + window partition
    ln1_window_kernel<<<NTOK / 8, 256, 0, stream>>>(x, ln1_g, ln1_b, A0);

    // 3) QKV GEMM  [32768,384] x [384,1152]
    gemm_bf16_kernel<<<dim3(1152 / 128, NTOK / 128), 256, 0, stream>>>(
        A0, qkv_wT, qkv_b, NTOK, 1152, 384, MODE_QKV, qbuf, nullptr, nullptr);

    // 4) windowed attention (512 windows x 8 heads)
    attn_kernel<<<NWIN * HEADS_, 256, 0, stream>>>(
        qbuf, qbuf + QK_ELEMS, qbuf + 2u * QK_ELEMS, attn_out);

    // 5) proj GEMM + window reverse + unshift + residual -> xh fp32
    gemm_bf16_kernel<<<dim3(384 / 128, NTOK / 128), 256, 0, stream>>>(
        attn_out, proj_wT, proj_b, NTOK, 384, 384, MODE_PROJ, nullptr, xh, x);

    // 6) LN2
    ln2_kernel<<<NTOK / 8, 256, 0, stream>>>(xh, ln2_g, ln2_b, hbuf);

    // 7) MLP fc1 + GELU  [32768,384] x [384,1536]
    gemm_bf16_kernel<<<dim3(1536 / 128, NTOK / 128), 256, 0, stream>>>(
        hbuf, w1T, mlp_b1, NTOK, 1536, 384, MODE_GELU, hid, nullptr, nullptr);

    // 8) MLP fc2 -> d_out [B,C,H,W] fp32
    gemm_bf16_kernel<<<dim3(384 / 128, NTOK / 128), 256, 0, stream>>>(
        hid, w2T, mlp_b2, NTOK, 384, 1536, MODE_OUT, nullptr, outp, nullptr);
}
